// VDEncoderDecoder_66709432041702
// MI455X (gfx1250) — matrix operand order repaired and checked by reading, not compiled, not run on hardware
//
#include <hip/hip_runtime.h>

// ---------------------------------------------------------------------------
// Fully-fused LSTM encoder-decoder for MI455X (gfx1250, wave32, WMMA).
//
// Workgroup: 128 threads = 4 wave32, owns 16 batch rows.
// All sequence state lives in LDS (~132 KB/WG, under the 320 KB WGP limit).
// Gate GEMMs use v_wmma_f32_16x16x32_f16 (f16 in, f32 accumulate).
// Activations use the CDNA5 hardware v_tanh_f32 transcendental.
// ---------------------------------------------------------------------------

typedef __attribute__((ext_vector_type(16))) _Float16 v16h;
typedef __attribute__((ext_vector_type(8)))  float    v8f;

union V16H { v16h v; unsigned u[8]; _Float16 h[16]; };
union V8F  { v8f  v; float f[8]; };

#define NTHREADS 128
#define ROWS 16            // batch rows per workgroup (== WMMA M)
#define KS   32            // K-padded stride of sequence buffers (halves)

// LDS carve (bytes)
//  bufA   : 60*16*32 f16  = 61440
//  bufB   : 60*16*32 f16  = 61440
//  s_gates: 16*128  f32   =  8192
//  s_c    : 16*32   f32   =  2048
//  s_h    : 16*32   f16   =  1024
//  s_bias : 128     f32   =   512
#define OFF_BUFA   0
#define OFF_BUFB   61440
#define OFF_GATES  122880
#define OFF_C      131072
#define OFF_H      133120
#define OFF_BIAS   134144
#define SMEM_BYTES 134656

// ---- fast activations: v_tanh_f32 when available, no IEEE divides --------
__device__ __forceinline__ float fast_tanh(float x) {
#if __has_builtin(__builtin_amdgcn_tanhf)
  return __builtin_amdgcn_tanhf(x);        // v_tanh_f32 (CDNA5 trans op)
#else
  // tanh(x) = 1 - 2/(exp(2x)+1), with v_rcp_f32 instead of IEEE divide
  const float e = __expf(2.0f * x);
  return 1.0f - 2.0f * __builtin_amdgcn_rcpf(e + 1.0f);
#endif
}

__device__ __forceinline__ float fast_sigmoid(float x) {
  // sigmoid(x) = 0.5*tanh(x/2) + 0.5 : one trans op + two VALU
  return __builtin_fmaf(0.5f, fast_tanh(0.5f * x), 0.5f);
}

// One LSTM layer over the 16-row tile.
//   gates(16 x 4H) = s_in_t(16 x IN) * w_ih^T  +  h(16 x H) * w_hh^T  + b
// IN <= 32, H <= 32 (single K=32 WMMA chunk each, zero padded).
// Gate column tiles of 16 are distributed across the 4 waves.
template<int IN, int H, int T>
__device__ void lstm_layer(const float* __restrict__ w_ih,
                           const float* __restrict__ w_hh,
                           const float* __restrict__ bias,
                           const _Float16* __restrict__ s_in,
                           _Float16* __restrict__ s_out,
                           float* s_gates, float* s_c, _Float16* s_h,
                           float* s_bias)
{
  const int tid  = threadIdx.x;
  const int lane = tid & 31;
  const int wave = tid >> 5;
  constexpr int NT  = (4 * H + 15) / 16;   // 16-wide gate tiles
  constexpr int TPW = (NT + 3) / 4;        // tiles per wave (max)

  // ---- layer init: zero out-seq (incl. K pad), h, c; stage bias ----------
  for (int i = tid; i < T * ROWS * (KS / 2); i += NTHREADS)
    ((unsigned*)s_out)[i] = 0u;
  for (int i = tid; i < ROWS * 32; i += NTHREADS) {
    s_c[i] = 0.0f;
    s_h[i] = (_Float16)0.0f;
  }
  for (int n = tid; n < 4 * H; n += NTHREADS) s_bias[n] = bias[n];

  // ---- build B operands (weights) in contiguous k order -----------------
  // B element (k, n): n = 16*nt + (lane&15); lanes 0-15 hold K=0..15 across
  // the 8 VGPR pairs, lanes 16-31 hold K=16..31.
  V16H Bih[TPW], Bhh[TPW];
  const int khiB = (lane >= 16) ? 16 : 0;
  #pragma unroll
  for (int i = 0; i < TPW; ++i) {
    const int  nt = wave + 4 * i;
    const int  n  = nt * 16 + (lane & 15);
    const bool nv = (nt < NT) && (n < 4 * H);
    #pragma unroll
    for (int p = 0; p < 16; ++p) {
      const int k = p + khiB;
      float wih = (nv && (k < IN)) ? w_ih[n * IN + k] : 0.0f;
      float whh = (nv && (k < H )) ? w_hh[n * H  + k] : 0.0f;
      Bih[i].h[p] = (_Float16)wih;
      Bhh[i].h[p] = (_Float16)whh;
    }
  }
  __syncthreads();

  const int m    = lane & 15;          // WMMA A row for this lane
  const int khiA = (lane >= 16) ? 8 : 0;

  for (int t = 0; t < T; ++t) {
    // ---- A operands in contiguous k order (as B) (packed f16 pairs from LDS) --
    // pair v covers K = {k0, k0+1}, k0 = 2v + (hi ? 16 : 0)
    V16H Ax, Ah;
    const unsigned* inrow = (const unsigned*)(s_in + (t * ROWS + m) * KS);
    const unsigned* hrow  = (const unsigned*)(s_h  + m * 32);
    #pragma unroll
    for (int v = 0; v < 8; ++v) {
      const int k0 = 2 * v + 2 * khiA;
      Ax.u[v] = inrow[k0 >> 1];
      Ah.u[v] = hrow[k0 >> 1];
    }

    // ---- gate GEMM: two chained WMMAs per tile ---------------------------
    #pragma unroll
    for (int i = 0; i < TPW; ++i) {
      const int nt = wave + 4 * i;
      if (nt < NT) {                    // uniform per wave
        v8f acc = {};
        acc = __builtin_amdgcn_wmma_f32_16x16x32_f16(
                  false, Ax.v, false, Bih[i].v, (short)0, acc, false, false);
        acc = __builtin_amdgcn_wmma_f32_16x16x32_f16(
                  false, Ah.v, false, Bhh[i].v, (short)0, acc, false, false);
        V8F d; d.v = acc;
        const int n = nt * 16 + m;      // D: reg r -> (row r+khiA, col n)
        #pragma unroll
        for (int r = 0; r < 8; ++r)
          s_gates[(r + khiA) * 128 + n] = d.f[r];
      }
    }
    __syncthreads();

    // ---- elementwise LSTM cell update (all 128 lanes) --------------------
    for (int idx = tid; idx < ROWS * H; idx += NTHREADS) {
      const int mm = idx / H, j = idx % H;   // H is a power of two
      const float* g = s_gates + mm * 128;
      const float iv = fast_sigmoid(g[0 * H + j] + s_bias[0 * H + j]);
      const float fv = fast_sigmoid(g[1 * H + j] + s_bias[1 * H + j]);
      const float gv = fast_tanh   (g[2 * H + j] + s_bias[2 * H + j]);
      const float ov = fast_sigmoid(g[3 * H + j] + s_bias[3 * H + j]);
      const float c  = fv * s_c[mm * 32 + j] + iv * gv;
      const float hv = ov * fast_tanh(c);
      s_c[mm * 32 + j] = c;
      const _Float16 h16 = (_Float16)hv;
      s_h[mm * 32 + j] = h16;                       // A-layout source
      s_out[(t * ROWS + mm) * KS + j] = h16;        // sequence output
    }
    __syncthreads();
  }
}

__global__ void __launch_bounds__(NTHREADS)
vd_encdec_kernel(const float* __restrict__ x,
                 const float* w_e1_ih, const float* w_e1_hh, const float* b_e1,
                 const float* w_e2_ih, const float* w_e2_hh, const float* b_e2,
                 const float* w_e3_ih, const float* w_e3_hh, const float* b_e3,
                 const float* w_d1_ih, const float* w_d1_hh, const float* b_d1,
                 const float* w_d2_ih, const float* w_d2_hh, const float* b_d2,
                 const float* w_d3_ih, const float* w_d3_hh, const float* b_d3,
                 const float* fc1_w, const float* fc1_b,
                 const float* fc2_w, const float* fc2_b,
                 float* __restrict__ out)
{
  extern __shared__ char smem[];
  _Float16* bufA    = (_Float16*)(smem + OFF_BUFA);
  _Float16* bufB    = (_Float16*)(smem + OFF_BUFB);
  float*    s_gates = (float*)   (smem + OFF_GATES);
  float*    s_c     = (float*)   (smem + OFF_C);
  _Float16* s_h     = (_Float16*)(smem + OFF_H);
  float*    s_bias  = (float*)   (smem + OFF_BIAS);

  const int  tid     = threadIdx.x;
  const long rowbase = (long)blockIdx.x * ROWS;

  // ---- load x tile: [16 rows][48 t][8 feat] -> bufA (f16, K-padded) ------
  for (int i = tid; i < 60 * ROWS * (KS / 2); i += NTHREADS)
    ((unsigned*)bufA)[i] = 0u;
  __syncthreads();
  for (int idx = tid; idx < ROWS * 48 * 8; idx += NTHREADS) {
    const int mm  = idx / (48 * 8);
    const int rem = idx % (48 * 8);
    const int t = rem >> 3, k = rem & 7;
    const float v = x[(rowbase + mm) * (48 * 8) + rem];
    bufA[(t * ROWS + mm) * KS + k] = (_Float16)v;
  }
  __syncthreads();

  // ---- encoder -----------------------------------------------------------
  lstm_layer< 8, 32, 48>(w_e1_ih, w_e1_hh, b_e1, bufA, bufB, s_gates, s_c, s_h, s_bias);
  lstm_layer<32,  8, 48>(w_e2_ih, w_e2_hh, b_e2, bufB, bufA, s_gates, s_c, s_h, s_bias);
  lstm_layer< 8,  1, 48>(w_e3_ih, w_e3_hh, b_e3, bufA, bufB, s_gates, s_c, s_h, s_bias);

  // ---- decoder input: concat(e3 out (48), x[:, -12:, col 4]) -> 60 steps -
  for (int i = tid; i < 60 * ROWS * (KS / 2); i += NTHREADS)
    ((unsigned*)bufA)[i] = 0u;
  __syncthreads();
  for (int idx = tid; idx < ROWS * 60; idx += NTHREADS) {
    const int mm = idx / 60, t = idx % 60;
    float v;
    if (t < 48) v = (float)bufB[(t * ROWS + mm) * KS + 0];
    else        v = x[(rowbase + mm) * (48 * 8) + (36 + (t - 48)) * 8 + 4];
    bufA[(t * ROWS + mm) * KS + 0] = (_Float16)v;
  }
  __syncthreads();

  lstm_layer<1, 2, 60>(w_d1_ih, w_d1_hh, b_d1, bufA, bufB, s_gates, s_c, s_h, s_bias);
  lstm_layer<2, 2, 60>(w_d2_ih, w_d2_hh, b_d2, bufB, bufA, s_gates, s_c, s_h, s_bias);
  lstm_layer<2, 1, 60>(w_d3_ih, w_d3_hh, b_d3, bufA, bufB, s_gates, s_c, s_h, s_bias);

  // ---- FC head: (B,60) @ fc1^T + b -> (B,32) @ fc2^T + b -> (B,12) -------
  for (int idx = tid; idx < ROWS * 32; idx += NTHREADS) {
    const int mm = idx >> 5, o = idx & 31;
    float acc = fc1_b[o];
    #pragma unroll 4
    for (int k = 0; k < 60; ++k)
      acc += (float)bufB[(k * ROWS + mm) * KS + 0] * fc1_w[o * 60 + k];
    s_gates[mm * 128 + o] = acc;   // reuse gate buffer for y1[16][32]
  }
  __syncthreads();
  for (int idx = tid; idx < ROWS * 12; idx += NTHREADS) {
    const int mm = idx / 12, o = idx % 12;
    float acc = fc2_b[o];
    #pragma unroll 4
    for (int k = 0; k < 32; ++k)
      acc += s_gates[mm * 128 + k] * fc2_w[o * 32 + k];
    out[(rowbase + mm) * 12 + o] = acc;
  }
}

extern "C" void kernel_launch(void* const* d_in, const int* in_sizes, int n_in,
                              void* d_out, int out_size, void* d_ws, size_t ws_size,
                              hipStream_t stream) {
  (void)in_sizes; (void)n_in; (void)d_ws; (void)ws_size; (void)out_size;

  const float* x       = (const float*)d_in[0];
  const float* w_e1_ih = (const float*)d_in[1];
  const float* w_e1_hh = (const float*)d_in[2];
  const float* b_e1    = (const float*)d_in[3];
  const float* w_e2_ih = (const float*)d_in[4];
  const float* w_e2_hh = (const float*)d_in[5];
  const float* b_e2    = (const float*)d_in[6];
  const float* w_e3_ih = (const float*)d_in[7];
  const float* w_e3_hh = (const float*)d_in[8];
  const float* b_e3    = (const float*)d_in[9];
  const float* w_d1_ih = (const float*)d_in[10];
  const float* w_d1_hh = (const float*)d_in[11];
  const float* b_d1    = (const float*)d_in[12];
  const float* w_d2_ih = (const float*)d_in[13];
  const float* w_d2_hh = (const float*)d_in[14];
  const float* b_d2    = (const float*)d_in[15];
  const float* w_d3_ih = (const float*)d_in[16];
  const float* w_d3_hh = (const float*)d_in[17];
  const float* b_d3    = (const float*)d_in[18];
  const float* fc1_w   = (const float*)d_in[19];
  const float* fc1_b   = (const float*)d_in[20];
  const float* fc2_w   = (const float*)d_in[21];
  const float* fc2_b   = (const float*)d_in[22];

  (void)hipFuncSetAttribute((const void*)vd_encdec_kernel,
                            hipFuncAttributeMaxDynamicSharedMemorySize,
                            SMEM_BYTES);

  const int nblocks = 32768 / ROWS;   // 2048 workgroups of 16 rows
  vd_encdec_kernel<<<nblocks, NTHREADS, SMEM_BYTES, stream>>>(
      x,
      w_e1_ih, w_e1_hh, b_e1,
      w_e2_ih, w_e2_hh, b_e2,
      w_e3_ih, w_e3_hh, b_e3,
      w_d1_ih, w_d1_hh, b_d1,
      w_d2_ih, w_d2_hh, b_d2,
      w_d3_ih, w_d3_hh, b_d3,
      fc1_w, fc1_b, fc2_w, fc2_b,
      (float*)d_out);
}
